// AdaATModule_47433618817210
// MI455X (gfx1250) — matrix operand order repaired and checked by reading, not compiled, not run on hardware
//
#include <hip/hip_runtime.h>
#include <math.h>

// ---------------------------------------------------------------------------
// AdaAT: tiny WMMA MLP head (one workgroup) + L2-gather-bound trilinear
// grid-sample body. MI455X / gfx1250, wave32.
//
// Roofline: head is ~2.6 MFLOP (negligible, done with v_wmma_f32_16x16x32_f16,
// one workgroup). Body is 16.8M outputs x 8 scattered b32 gathers over a 67 MB
// feature map that fits the 192 MB L2 -> L2-gather-bound. Output stores are
// write-once, so they use the NT temporal hint to avoid evicting the gather
// working set from L2.
// ---------------------------------------------------------------------------

typedef _Float16 half_t;
typedef __attribute__((ext_vector_type(16))) _Float16 v16h;
typedef __attribute__((ext_vector_type(8)))  float    v8f;

#define BATCH 4
#define PARA  256
#define FEAT  256
#define PI_F  3.14159f

// ===========================================================================
// Kernel 1: parameter head.
//   p     = relu(para @ W1 + b1)                [4,256]  (M padded to 16)
//   scale = sigmoid(p @ Ws + bs) * 2            [4,256]
//   angle = tanh(p @ Wr + br) * PI  -> cos,sin  [4,256]
//   trans = tanh(p @ Wt + bt)                   [4,512] -> tx,ty [4,256]
// 16 waves (512 threads), one workgroup. All GEMMs via v_wmma_f32_16x16x32_f16.
// ===========================================================================
__launch_bounds__(512)
__global__ void adaat_params_kernel(const float* __restrict__ para,
                                    const float* __restrict__ W1, const float* __restrict__ b1,
                                    const float* __restrict__ Ws, const float* __restrict__ bs,
                                    const float* __restrict__ Wr, const float* __restrict__ br,
                                    const float* __restrict__ Wt, const float* __restrict__ bt,
                                    float* __restrict__ pCa, float* __restrict__ pSa,
                                    float* __restrict__ pSc, float* __restrict__ pTx,
                                    float* __restrict__ pTy) {
    __shared__ half_t ldsX[16 * PARA];   // padded input activations (rows 4..15 = 0)
    __shared__ half_t ldsP[16 * PARA];   // hidden activations p

    const int tid    = threadIdx.x;
    const int wave   = tid >> 5;
    const int lane   = tid & 31;
    const int l15    = lane & 15;
    const int laneHi = lane >> 4;        // 0: lanes 0-15, 1: lanes 16-31

    // Load X (f32 -> f16), zero-pad rows 4..15.
    for (int i = tid; i < 16 * PARA; i += 512) {
        int row = i >> 8, k = i & 255;
        ldsX[i] = (row < BATCH) ? (half_t)para[row * PARA + k] : (half_t)0.0f;
    }
    __syncthreads();

    // ---- Stage 1: hidden layer. Wave w owns output columns [16w, 16w+16). ----
    {
        const int nb  = wave * 16;
        const int col = nb + l15;
        v8f acc = {};
        for (int kt = 0; kt < PARA / 32; ++kt) {
            v16h a, bf;
#pragma unroll
            for (int v = 0; v < 8; ++v) {
                // A 16x32 f16 layout: VGPR v<4 -> K = 2v (+8 for upper lanes),
                //                     VGPR v>=4 -> K = 16 + 2(v-4) (+8 upper).
                int kb = (v < 4) ? (2 * v + laneHi * 8) : (16 + 2 * (v - 4) + laneHi * 8);
                int k  = kt * 32 + kb;
                a[2 * v]     = ldsX[l15 * PARA + k];
                a[2 * v + 1] = ldsX[l15 * PARA + k + 1];
            }
#pragma unroll
            for (int v = 0; v < 8; ++v) {
                // B 32x16 f16: lane = N column; same K order as A (+8 for upper lanes).
                int k = kt * 32 + ((v < 4) ? (2 * v + laneHi * 8) : (16 + 2 * (v - 4) + laneHi * 8));
                bf[2 * v]     = (half_t)W1[k * PARA + col];
                bf[2 * v + 1] = (half_t)W1[(k + 1) * PARA + col];
            }
            acc = __builtin_amdgcn_wmma_f32_16x16x32_f16(false, a, false, bf,
                                                         (short)0, acc, false, false);
        }
        const float bias = b1[col];
#pragma unroll
        for (int r = 0; r < 8; ++r) {
            int   m = r + laneHi * 8;            // C/D layout: M = r + 8*laneHi
            float v = acc[r] + bias;
            ldsP[m * PARA + col] = (half_t)(v > 0.0f ? v : 0.0f);
        }
    }
    __syncthreads();

    // ---- Stage 2: concatenated [Ws(256) | Wr(256) | Wt(512)] = 64 tiles. ----
    for (int t = 0; t < 4; ++t) {
        const int tile = wave * 4 + t;
        const int j0   = tile * 16;
        const float* Wm;  const float* bias;  int ldN, jloc;
        if (j0 < 256)      { Wm = Ws; bias = bs; ldN = FEAT;     jloc = j0;       }
        else if (j0 < 512) { Wm = Wr; bias = br; ldN = FEAT;     jloc = j0 - 256; }
        else               { Wm = Wt; bias = bt; ldN = 2 * FEAT; jloc = j0 - 512; }

        const int col = jloc + l15;
        v8f acc = {};
        for (int kt = 0; kt < PARA / 32; ++kt) {
            v16h a, bf;
#pragma unroll
            for (int v = 0; v < 8; ++v) {
                int kb = (v < 4) ? (2 * v + laneHi * 8) : (16 + 2 * (v - 4) + laneHi * 8);
                int k  = kt * 32 + kb;
                a[2 * v]     = ldsP[l15 * PARA + k];
                a[2 * v + 1] = ldsP[l15 * PARA + k + 1];
            }
#pragma unroll
            for (int v = 0; v < 8; ++v) {
                int k = kt * 32 + ((v < 4) ? (2 * v + laneHi * 8) : (16 + 2 * (v - 4) + laneHi * 8));
                bf[2 * v]     = (half_t)Wm[k * ldN + col];
                bf[2 * v + 1] = (half_t)Wm[(k + 1) * ldN + col];
            }
            acc = __builtin_amdgcn_wmma_f32_16x16x32_f16(false, a, false, bf,
                                                         (short)0, acc, false, false);
        }
        const float bv = bias[col];
#pragma unroll
        for (int r = 0; r < 8; ++r) {
            int m = r + laneHi * 8;
            if (m >= BATCH) continue;            // only rows 0..3 are real batches
            float v = acc[r] + bv;
            if (j0 < 256) {
                pSc[m * FEAT + col] = 2.0f / (1.0f + expf(-v));
            } else if (j0 < 512) {
                float ang = tanhf(v) * PI_F;
                pCa[m * FEAT + col] = cosf(ang);
                pSa[m * FEAT + col] = sinf(ang);
            } else {
                float tv  = tanhf(v);
                int   cch = col >> 1;            // translation reshape [B,C,2]
                if (col & 1) pTy[m * FEAT + cch] = tv;
                else         pTx[m * FEAT + cch] = tv;
            }
        }
    }
}

// ===========================================================================
// Kernel 2: affine grid + trilinear sample (padding zeros, align_corners=False
// sampling of an align_corners=True grid, exactly as the reference).
// One thread per output pixel; each block stays inside one (b,c) plane so the
// five affine params and the whole z-interpolation are block-uniform (SGPR).
// 8 clamped b32 gathers per output -> L2-resident gather stream. Output
// stores are non-temporal so the write-once result does not evict the
// feature-map working set from the 192 MB L2.
// ===========================================================================
__launch_bounds__(256)
__global__ void adaat_sample_kernel(const float* __restrict__ fm,
                                    const float* __restrict__ pCa,
                                    const float* __restrict__ pSa,
                                    const float* __restrict__ pSc,
                                    const float* __restrict__ pTx,
                                    const float* __restrict__ pTy,
                                    float* __restrict__ out) {
    const int C = FEAT, H = 128, W = 128;
    const int b  = blockIdx.z;
    const int c  = blockIdx.y;
    const int bc = b * C + c;

    // Block-uniform affine parameters (scalar loads).
    const float ca = pCa[bc], sa = pSa[bc], sc = pSc[bc];
    const float tx = pTx[bc], ty = pTy[bc];

    // Block-uniform z interpolation: gz = 2*c/(C-1)-1; iz = ((gz+1)*D-1)/2.
    const float gz = (float)c * (2.0f / 255.0f) - 1.0f;
    const float iz = gz * 128.0f + 127.5f;
    const float zf = floorf(iz);
    const float wz = iz - zf;
    const int   z0 = (int)zf;
    const int   z1 = z0 + 1;

    const int pix = blockIdx.x * blockDim.x + threadIdx.x;   // 0..16383
    const int h   = pix >> 7;
    const int w   = pix & 127;

    // align_corners=True style grid coords.
    const float x = (float)w * (2.0f / 127.0f) - 1.0f;
    const float y = (float)h * (2.0f / 127.0f) - 1.0f;

    // rot @ [x,y] * scale + trans
    const float gx = (ca * x - sa * y) * sc + tx;
    const float gy = (sa * x + ca * y) * sc + ty;

    // grid_sample index math (align_corners=False): i = ((g+1)*N-1)/2
    const float ix = gx * 64.0f + 63.5f;
    const float iy = gy * 64.0f + 63.5f;
    const float xf = floorf(ix), yf = floorf(iy);
    const float wx = ix - xf,    wy = iy - yf;
    const int x0 = (int)xf, x1 = x0 + 1;
    const int y0 = (int)yf, y1 = y0 + 1;

    const float* __restrict__ vol = fm + (size_t)b * C * H * W;

    auto samp = [&](int zz, int yy, int xx) -> float {
        bool ok = ((unsigned)zz < 256u) & ((unsigned)yy < 128u) & ((unsigned)xx < 128u);
        int zc = min(max(zz, 0), 255);
        int yc = min(max(yy, 0), 127);
        int xc = min(max(xx, 0), 127);
        float v = vol[((zc << 7) + yc) * 128 + xc];
        return ok ? v : 0.0f;
    };

    const float v000 = samp(z0, y0, x0), v001 = samp(z0, y0, x1);
    const float v010 = samp(z0, y1, x0), v011 = samp(z0, y1, x1);
    const float v100 = samp(z1, y0, x0), v101 = samp(z1, y0, x1);
    const float v110 = samp(z1, y1, x0), v111 = samp(z1, y1, x1);

    const float c00 = v000 + wx * (v001 - v000);
    const float c01 = v010 + wx * (v011 - v010);
    const float c10 = v100 + wx * (v101 - v100);
    const float c11 = v110 + wx * (v111 - v110);
    const float c0  = c00 + wy * (c01 - c00);
    const float c1  = c10 + wy * (c11 - c10);
    const float r   = c0 + wz * (c1 - c0);

    // Non-temporal store: stream write-once output past L2 residency.
    __builtin_nontemporal_store(r, &out[(((size_t)bc) * H + h) * W + w]);
}

// ===========================================================================
extern "C" void kernel_launch(void* const* d_in, const int* in_sizes, int n_in,
                              void* d_out, int out_size, void* d_ws, size_t ws_size,
                              hipStream_t stream) {
    (void)in_sizes; (void)n_in; (void)out_size; (void)ws_size;

    const float* fm   = (const float*)d_in[0];   // [4,256,128,128]
    const float* para = (const float*)d_in[1];   // [4,256]
    const float* W1   = (const float*)d_in[2];   // [256,256]
    const float* b1   = (const float*)d_in[3];   // [256]
    const float* Ws   = (const float*)d_in[4];   // [256,256]
    const float* bs   = (const float*)d_in[5];   // [256]
    const float* Wr   = (const float*)d_in[6];   // [256,256]
    const float* br   = (const float*)d_in[7];   // [256]
    const float* Wt   = (const float*)d_in[8];   // [256,512]
    const float* bt   = (const float*)d_in[9];   // [512]
    float* out = (float*)d_out;

    // Parameter scratch in d_ws: 5 arrays of B*C floats (20 KB).
    float* pCa = (float*)d_ws;
    float* pSa = pCa + BATCH * FEAT;
    float* pSc = pSa + BATCH * FEAT;
    float* pTx = pSc + BATCH * FEAT;
    float* pTy = pTx + BATCH * FEAT;

    adaat_params_kernel<<<1, 512, 0, stream>>>(para, W1, b1, Ws, bs, Wr, br, Wt, bt,
                                               pCa, pSa, pSc, pTx, pTy);

    dim3 grid((128 * 128) / 256, FEAT, BATCH);   // (64, 256, 4)
    adaat_sample_kernel<<<grid, 256, 0, stream>>>(fm, pCa, pSa, pSc, pTx, pTy, out);
}
